// CosSim2d_9208409882745
// MI455X (gfx1250) — matrix operand order repaired and checked by reading, not compiled, not run on hardware
//
#include <hip/hip_runtime.h>
#include <math.h>

typedef __attribute__((ext_vector_type(16))) _Float16 v16h;
typedef __attribute__((ext_vector_type(8)))  float    v8f;
typedef __attribute__((ext_vector_type(4)))  float    v4f;

#define CS_EPS 1e-12f

// ---------------------------------------------------------------------------
// Kernel 1: normalize weights and emit them in WMMA B-fragment layout.
// frag layout: [vt(4)][step(18)][lane(32)][j(16)]  (f16)
// Dense B 32x16 layout used here (contiguous k, same as A): n = lane&15, K = j + 16*(lane>>4)
// ---------------------------------------------------------------------------
__global__ void wn_frag_kernel(const float* __restrict__ w,
                               _Float16* __restrict__ frag) {
    int t = blockIdx.x * blockDim.x + threadIdx.x;
    if (t >= 4 * 18 * 32 * 16) return;
    int j    = t & 15;
    int lane = (t >> 4) & 31;
    int s    = (t >> 9) % 18;
    int vt   = t / (18 * 512);
    int half = lane >> 4;
    int v    = vt * 16 + (lane & 15);
    int k    = s * 32 + j + 16 * half;      // B-fragment K mapping
    int c    = k / 9;
    int l    = k - c * 9;
    const float* wp = w + (v * 64 + c) * 9;
    float sum = 0.f;
#pragma unroll
    for (int i = 0; i < 9; ++i) sum += wp[i] * wp[i];
    float rn = 1.0f / fmaxf(sqrtf(sum), CS_EPS);
    frag[t] = (_Float16)(wp[l] * rn);
}

// ---------------------------------------------------------------------------
// Kernel 2: cosine-sim GEMM via v_wmma_f32_16x16x32_f16.
// Block = 128 threads = 4 waves. Block handles 16 w-positions of one (n,h)
// row across all 64 couts (wave wv -> couts [16*wv, 16*wv+16)).
// M-tile rows m = 0..15 map to w = w0+m ; K = c*9 + tap, K total = 576.
// ---------------------------------------------------------------------------
__global__ __launch_bounds__(128)
void cossim_kernel(const float* __restrict__ x,
                   const _Float16* __restrict__ bfrag,
                   const float* __restrict__ p,
                   float* __restrict__ out) {
    __shared__ float xs[64 * 3 * 18];              // [c][dy][col], halo, zero-padded
    __shared__ float rnorm[64 * 16];               // [c][m] inverse patch norms
    __shared__ __align__(32) _Float16 As[18 * 32 * 16]; // A fragments [step][lane][j]

    const int t    = threadIdx.x;
    const int lane = t & 31;
    const int wv   = t >> 5;
    const int bid  = blockIdx.x;
    const int w0   = (bid & 3) * 16;
    const int h    = (bid >> 2) & 63;
    const int n    = bid >> 8;

    // ---- Phase 1: stage zero-padded 64ch x 3row x 18col halo into LDS ----
    const float* xb = x + n * (64 * 64 * 64);
    for (int e = t; e < 64 * 3 * 18; e += 128) {
        int col  = e % 18;
        int rest = e / 18;
        int dy   = rest % 3;
        int c    = rest / 3;
        int row  = h + dy - 1;
        int gcol = w0 + col - 1;
        float v = 0.f;
        if (row >= 0 && row < 64 && gcol >= 0 && gcol < 64)
            v = xb[(c * 64 + row) * 64 + gcol];
        xs[e] = v;
    }
    __syncthreads();

    // ---- Phase 2: inverse L2 norm of each (c, position) 3x3 patch ----
    for (int e = t; e < 64 * 16; e += 128) {
        int c = e >> 4, m = e & 15;
        const float* xc = xs + c * 54;
        float sum = 0.f;
#pragma unroll
        for (int dy = 0; dy < 3; ++dy)
#pragma unroll
            for (int dx = 0; dx < 3; ++dx) {
                float v = xc[dy * 18 + m + dx];
                sum += v * v;
            }
        rnorm[e] = 1.0f / fmaxf(sqrtf(sum), CS_EPS);
    }
    __syncthreads();

    // ---- Phase 3: build normalized A fragments once, shared by all 4 waves.
    // Dense A 16x32 layout used here (contiguous k, same as B): m = lane&15, K = j + 16*half.
    for (int e = t; e < 18 * 32 * 16; e += 128) {
        int j    = e & 15;
        int ln   = (e >> 4) & 31;
        int s    = e >> 9;
        int half = ln >> 4;
        int m    = ln & 15;
        int k    = s * 32 + j + 16 * half;
        int c    = k / 9;
        int l    = k - c * 9;
        int dy   = l / 3, dx = l - dy * 3;
        float v = xs[c * 54 + dy * 18 + m + dx] * rnorm[c * 16 + m];
        As[e] = (_Float16)v;
    }
    __syncthreads();

    // ---- Phase 4: K = 576 -> 18 WMMA steps of 32 ----
    v8f acc = {};
    const _Float16* bw = bfrag + wv * (18 * 512);
#pragma unroll 3
    for (int s = 0; s < 18; ++s) {
        v16h a = *(const v16h*)(As + (s * 32 + lane) * 16);   // ds_load_b128 x2
        v16h b = *(const v16h*)(bw + (s * 32 + lane) * 16);   // global_load_b128 x2 (L2)
        acc = __builtin_amdgcn_wmma_f32_16x16x32_f16(
            /*neg_a=*/false, a, /*neg_b=*/false, b,
            /*c_mod=*/(short)0, acc, /*reuse_a=*/false, /*reuse_b=*/false);
    }

    // ---- Epilogue: sign-preserving pow via raw v_log_f32 / v_exp_f32 ----
    // base = |s|+eps > 0 always, so no pow special cases are reachable:
    //   mag = exp2(p * log2(base));  out = (s==0) ? 0 : copysign(mag, s)
    // C/D layout: lane -> N (cout within tile), VGPR r -> M = r + 8*(lane>>4).
    const int half = lane >> 4;
    const int v    = wv * 16 + (lane & 15);
    const float pv = p[v];
    float res[8];
#pragma unroll
    for (int r = 0; r < 8; ++r) {
        float s   = acc[r];
        float lg  = __builtin_amdgcn_logf(fabsf(s) + CS_EPS);   // v_log_f32 (log2)
        float mag = __builtin_amdgcn_exp2f(pv * lg);            // v_exp_f32 (2^x)
        res[r] = (s == 0.f) ? 0.f : __builtin_copysignf(mag, s);
    }
    float* o = out + ((n * 64 + v) * 64 + h) * 64 + w0 + half * 8;
    v4f o0 = {res[0], res[1], res[2], res[3]};
    v4f o1 = {res[4], res[5], res[6], res[7]};
    *(v4f*)(o)     = o0;
    *(v4f*)(o + 4) = o1;
}

// ---------------------------------------------------------------------------
extern "C" void kernel_launch(void* const* d_in, const int* in_sizes, int n_in,
                              void* d_out, int out_size, void* d_ws, size_t ws_size,
                              hipStream_t stream) {
    const float* x = (const float*)d_in[0];   // (32, 64, 64, 64) f32
    const float* w = (const float*)d_in[1];   // (64, 64, 9) f32
    const float* p = (const float*)d_in[2];   // (64,) f32
    float* out = (float*)d_out;               // (32, 64, 64, 64) f32
    _Float16* frag = (_Float16*)d_ws;         // 4*18*32*16 f16 = 73,728 B

    wn_frag_kernel<<<(4 * 18 * 32 * 16 + 255) / 256, 256, 0, stream>>>(w, frag);
    cossim_kernel<<<32 * 64 * 4, 128, 0, stream>>>(x, frag, p, out);
}
